// flowGAN_48438641164944
// MI455X (gfx1250) — matrix operand order repaired and checked by reading, not compiled, not run on hardware
//
#include <hip/hip_runtime.h>
#include <cmath>

typedef __attribute__((ext_vector_type(16))) _Float16 v16h;
typedef __attribute__((ext_vector_type(2)))  _Float16 h2;
typedef __attribute__((ext_vector_type(8)))  float    v8f;

#define NSTEP 8
#define HDIM  512
#define BLOCK 256
#define NWAVE (BLOCK / 32)

__device__ __forceinline__ float lane_bcast_f32(float v, int srcLane) {
    int r = __builtin_amdgcn_ds_bpermute(srcLane << 2, __builtin_bit_cast(int, v));
    return __builtin_bit_cast(float, r);
}

__global__ __launch_bounds__(BLOCK)
void flow_wmma_kernel(const float* __restrict__ x,
                      const float* __restrict__ Ws,
                      const float* __restrict__ an_logs,
                      const float* __restrict__ an_b,
                      const float* __restrict__ cw1,
                      const float* __restrict__ cb1,
                      const float* __restrict__ cw2,
                      const float* __restrict__ cb2,
                      float* __restrict__ out,
                      int B)
{
    // f16 copies of the coupling-net weights, laid out for paired (pk) access
    __shared__ h2 w1h[NSTEP * (HDIM / 2)];          // 8 KB
    __shared__ h2 b1h[NSTEP * (HDIM / 2)];          // 8 KB
    __shared__ h2 w2h[NSTEP * 2 * (HDIM / 2)];      // 16 KB
    __shared__ h2 zpad[8];                          // 32 B of zeros (B-frag pad)
    __shared__ float stS[NWAVE][2][2][16];          // [wave][tile][n][m], 2 KB

    const int tid  = threadIdx.x;
    const int lane = tid & 31;
    const int wave = tid >> 5;
    const int nidx = lane & 15;   // column / row-in-tile index
    const int hi   = lane >> 4;   // which half of the wave

    __builtin_prefetch(cw1, 0, 3);
    __builtin_prefetch(cw2, 0, 3);

    // cooperative f32 -> f16 preload into LDS
    _Float16* w1f = (_Float16*)w1h;
    _Float16* b1f = (_Float16*)b1h;
    _Float16* w2f = (_Float16*)w2h;
    for (int i = tid; i < NSTEP * HDIM; i += BLOCK) {
        w1f[i] = (_Float16)cw1[i];
        b1f[i] = (_Float16)cb1[i];
    }
    for (int i = tid; i < NSTEP * 2 * HDIM; i += BLOCK) {
        w2f[i] = (_Float16)cw2[i];
    }
    if (tid < 8) { h2 z; z.x = (_Float16)0.f; z.y = (_Float16)0.f; zpad[tid] = z; }
    __syncthreads();

    const int sidx = blockIdx.x * BLOCK + tid;
    const int ld   = (sidx < B) ? sidx : (B - 1);
    const float x0 = x[ld * 2 + 0];
    const float x1 = x[ld * 2 + 1];

    // initial LU mix with Ws[0]  (y = x @ W^T)
    float W00 = Ws[0], W01 = Ws[1], W10 = Ws[2], W11 = Ws[3];
    float y0 = x0 * W00 + x1 * W01;
    float y1 = x0 * W10 + x1 * W11;
    float lJ = logf(fabsf(W00 * W11 - W01 * W10));

    for (int step = 0; step < NSTEP; ++step) {
        // ---- actnorm ----
        float l0 = an_logs[step * 2 + 0], l1 = an_logs[step * 2 + 1];
        y0 = y0 * expf(l0) + an_b[step * 2 + 0];
        y1 = y1 * expf(l1) + an_b[step * 2 + 1];
        lJ += l0 + l1;

        // ---- coupling net: st(16x2) = relu(y0*w1+b1)(16x512) @ w2^T via WMMA ----
        // two 16-row tiles per wave: rows = wave-local samples 0..15 and 16..31
        float y0a = lane_bcast_f32(y0, nidx);
        float y0b = lane_bcast_f32(y0, nidx + 16);
        h2 y0ap; y0ap.x = (_Float16)y0a; y0ap.y = (_Float16)y0a;
        h2 y0bp; y0bp.x = (_Float16)y0b; y0bp.y = (_Float16)y0b;

        v8f acc0, acc1;
        #pragma unroll
        for (int r = 0; r < 8; ++r) { acc0[r] = 0.f; acc1[r] = 0.f; }

        const int pAstep = step * (HDIM / 2);
        // branchless B-fragment source: columns >= 2 read a zero pad region
        const h2* const bbase = (nidx < 2)
            ? (w2h + (step * 2 + nidx) * (HDIM / 2) + hi * 8)
            : zpad;
        const int bstride = (nidx < 2) ? 16 : 0;   // advance per kc only for real cols

        #pragma unroll 2
        for (int kc = 0; kc < HDIM / 32; ++kc) {
            // B fragment: w2^T chunk, 32(K) x 16(N); only N=0,1 nonzero.
            // Layout: lanes 0-15 hold K=0..15, lanes 16-31 hold K=16..31 (seq halves).
            const h2* bsrc = bbase + kc * bstride;
            v16h bf;
            #pragma unroll
            for (int p = 0; p < 8; ++p) {
                h2 wv = bsrc[p];
                bf[2 * p]     = wv.x;
                bf[2 * p + 1] = wv.y;
            }

            // A fragments: h = relu(y0*w1 + b1), 16(M) x 32(K).
            // Lane<16 holds K = 0..15; lane>=16 holds K = 16..31.
            int pa = pAstep + kc * 16 + hi * 8;
            v16h af0, af1;
            h2 z; z.x = (_Float16)0.f; z.y = (_Float16)0.f;
            #pragma unroll
            for (int p = 0; p < 8; ++p) {
                int pp = pa + p;  // pairs (0..15)
                h2 wv = w1h[pp];
                h2 bv = b1h[pp];
                h2 h0 = y0ap * wv + bv;               // v_pk_fma_f16
                h2 h1 = y0bp * wv + bv;
                h0 = __builtin_elementwise_max(h0, z);
                h1 = __builtin_elementwise_max(h1, z);
                af0[2 * p] = h0.x; af0[2 * p + 1] = h0.y;
                af1[2 * p] = h1.x; af1[2 * p + 1] = h1.y;
            }

            acc0 = __builtin_amdgcn_wmma_f32_16x16x32_f16(
                false, af0, false, bf, (short)0, acc0, false, false);
            acc1 = __builtin_amdgcn_wmma_f32_16x16x32_f16(
                false, af1, false, bf, (short)0, acc1, false, false);
        }

        // D layout: lane(n<16) holds column N=n, rows M=0..7 (V0..7); lane 16+n rows 8..15.
        if (nidx < 2) {
            #pragma unroll
            for (int r = 0; r < 8; ++r) {
                stS[wave][0][nidx][hi * 8 + r] = acc0[r];
                stS[wave][1][nidx][hi * 8 + r] = acc1[r];
            }
        }
        asm volatile("s_wait_dscnt 0" ::: "memory");  // wave-local LDS fence (LDS in-order per wave)

        float st0 = stS[wave][hi][0][nidx] + cb2[step * 2 + 0];
        float st1 = stS[wave][hi][1][nidx] + cb2[step * 2 + 1];

        float log_s = tanhf(st0);
        y1 = y1 * expf(log_s) + st1;
        lJ += log_s;

        // ---- LU mix with Ws[step+1] ----
        const float* W = Ws + (step + 1) * 4;
        float a = W[0], b_ = W[1], c = W[2], d = W[3];
        float ny0 = a * y0 + b_ * y1;
        float ny1 = c * y0 + d * y1;
        lJ += logf(fabsf(a * d - b_ * c));
        y0 = ny0; y1 = ny1;
    }

    if (sidx < B) {
        out[sidx * 2 + 0] = y0;
        out[sidx * 2 + 1] = y1;
        // log(1/(2*pi)) - ||y||^2/2 + lJ
        out[2 * B + sidx] = -1.8378770664093453f - 0.5f * (y0 * y0 + y1 * y1) + lJ;
    }
}

extern "C" void kernel_launch(void* const* d_in, const int* in_sizes, int n_in,
                              void* d_out, int out_size, void* d_ws, size_t ws_size,
                              hipStream_t stream) {
    const float* x       = (const float*)d_in[0];
    const float* Ws      = (const float*)d_in[1];
    const float* an_logs = (const float*)d_in[2];
    const float* an_b    = (const float*)d_in[3];
    const float* cw1     = (const float*)d_in[4];
    const float* cb1     = (const float*)d_in[5];
    const float* cw2     = (const float*)d_in[6];
    const float* cb2     = (const float*)d_in[7];

    const int B = in_sizes[0] / 2;
    dim3 grid((B + BLOCK - 1) / BLOCK);
    flow_wmma_kernel<<<grid, BLOCK, 0, stream>>>(
        x, Ws, an_logs, an_b, cw1, cb1, cw2, cb2, (float*)d_out, B);
}
